// DynamicPTTopicModeling_73280732004883
// MI455X (gfx1250) — compile-verified, matrix operand order checked by reading, not run on hardware
//
#include <hip/hip_runtime.h>

typedef __attribute__((ext_vector_type(16))) __bf16 v16bf;
typedef __attribute__((ext_vector_type(8)))  __bf16 v8bf;
typedef __attribute__((ext_vector_type(4)))  __bf16 v4bf;
typedef __attribute__((ext_vector_type(8)))  float  v8f;

constexpr int B_  = 8;
constexpr int C_  = 16;
constexpr int P_  = 512;
constexpr int D_  = 1024;   // feature dim
constexpr int G_  = 512;    // group dim
constexpr int ROWS = C_ * P_;        // 8192 rows per batch
constexpr int M_TILE = 64;           // rows per workgroup
constexpr int THREADS = 256;         // 8 waves: 4 strips x 2 N-halves
constexpr int APITCH = D_ + 8;       // bf16 elems; +16B pad to break bank-conflict alignment
constexpr int QPITCH = G_ + 8;

// ---------------------------------------------------------------------------
// Pre-pass: W (f32 [B,G,D]) -> Wb bf16 [B,G,D] and WTb bf16 [B,D,G]
// ---------------------------------------------------------------------------
__global__ void prep_weights(const float* __restrict__ W,
                             __bf16* __restrict__ Wb,
                             __bf16* __restrict__ WTb) {
  int i = blockIdx.x * blockDim.x + threadIdx.x;
  if (i >= B_ * G_ * D_) return;
  int d = i & (D_ - 1);
  int g = (i >> 10) & (G_ - 1);
  int b = i >> 19;
  __bf16 h = (__bf16)W[i];
  Wb[i] = h;
  WTb[((size_t)b * D_ + d) * G_ + g] = h;
}

// ---------------------------------------------------------------------------
// Fused: qg = relu(qz @ W^T); inv = 1/max(rowsum(qg),eps); msg = (qg @ W)*inv
// 8 waves: strip = wave&3 owns 16 rows; nhalf = wave>>2 splits the N loops so
// two waves per SIMD32 cover each other's memory latency.
// ---------------------------------------------------------------------------
__global__ __launch_bounds__(THREADS, 1)
void topic_fused(const float*  __restrict__ qz,
                 const __bf16* __restrict__ Wb,
                 const __bf16* __restrict__ WTb,
                 float*        __restrict__ out) {
  extern __shared__ char smem[];
  __bf16* a_lds  = (__bf16*)smem;                       // [M_TILE][APITCH]
  __bf16* qg_lds = a_lds + M_TILE * APITCH;             // [M_TILE][QPITCH]
  float*  sum_lds = (float*)(qg_lds + M_TILE * QPITCH); // [4 strips][2 halves][16 rows]

  const int mt   = blockIdx.x;       // row tile within batch
  const int bb   = blockIdx.y;       // batch
  const int tid  = threadIdx.x;
  const int wave  = tid >> 5;
  const int strip = wave & 3;              // which 16-row strip
  const int nhalf = wave >> 2;             // which half of the N loops
  const int lane  = tid & 31;
  const int lrow  = lane & 15;             // row (A) / col (B,C,D) within 16
  const int khalf = (lane >> 4) << 3;      // 0 or 8: K-chunk select, also C/D row offset

  // ---- Stage qz tile (64 x 1024) f32 -> bf16 LDS, fully cooperative ----
  const float* qbase = qz + ((size_t)bb * ROWS + (size_t)mt * M_TILE) * D_;
  for (int r = 0; r < M_TILE; ++r) {
    const float4* rowp = (const float4*)(qbase + (size_t)r * D_);
    float4 v0 = rowp[tid];                 // 256 threads x float4 = one 1024-col row
    v4bf p0 = {(__bf16)v0.x, (__bf16)v0.y, (__bf16)v0.z, (__bf16)v0.w};
    *(v4bf*)(a_lds + r * APITCH + 4 * tid) = p0;
  }
  __syncthreads();

  // Per-lane A base: lane holds row (lane&15), K-chunks {khalf..+7, khalf+16..+23}
  const __bf16* a_base  = a_lds + (strip * 16 + lrow) * APITCH + khalf;
  const __bf16* wb_base = Wb + (size_t)bb * G_ * D_ + (size_t)lrow * D_ + khalf;

  // ---- GEMM1 (this wave: 16 of 32 g-tiles) ----
  {
    float sums[8] = {0.f, 0.f, 0.f, 0.f, 0.f, 0.f, 0.f, 0.f};
    for (int i = 0; i < 16; ++i) {
      const int nt = nhalf * 16 + i;
      const __bf16* b_base = wb_base + (size_t)(nt * 16) * D_;
      v8f acc = {0.f, 0.f, 0.f, 0.f, 0.f, 0.f, 0.f, 0.f};
#pragma unroll 4
      for (int ks = 0; ks < D_ / 32; ++ks) {
        const __bf16* ap = a_base + ks * 32;
        const __bf16* bp = b_base + ks * 32;
        v8bf alo = *(const v8bf*)(ap);
        v8bf ahi = *(const v8bf*)(ap + 16);
        v8bf blo = *(const v8bf*)(bp);
        v8bf bhi = *(const v8bf*)(bp + 16);
        v16bf Am = __builtin_shufflevector(alo, ahi, 0,1,2,3,4,5,6,7,8,9,10,11,12,13,14,15);
        v16bf Bm = __builtin_shufflevector(blo, bhi, 0,1,2,3,4,5,6,7,8,9,10,11,12,13,14,15);
        acc = __builtin_amdgcn_wmma_f32_16x16x32_bf16(false, Am, false, Bm,
                                                      (short)0, acc, false, false);
      }
      // relu + partial L1 row sums + store unnormalized bf16 qg to LDS.
      // C/D layout: lane L, vgpr j -> row j + khalf, col nt*16 + (L&15)
      __bf16* qdst = qg_lds + (strip * 16 + khalf) * QPITCH + nt * 16 + lrow;
#pragma unroll
      for (int j = 0; j < 8; ++j) {
        float v = fmaxf(acc[j], 0.0f);
        sums[j] += v;
        qdst[j * QPITCH] = (__bf16)v;
      }
    }
    // Reduce row sums: xor masks 1..8 stay within each 16-lane half (wave32),
    // then publish this wave's partial sums (lane 0 -> rows 0..7, lane 16 -> 8..15)
#pragma unroll
    for (int j = 0; j < 8; ++j) {
      float s = sums[j];
      s += __shfl_xor(s, 1, 32);
      s += __shfl_xor(s, 2, 32);
      s += __shfl_xor(s, 4, 32);
      s += __shfl_xor(s, 8, 32);
      sums[j] = s;
    }
    if (lrow == 0) {
#pragma unroll
      for (int j = 0; j < 8; ++j)
        sum_lds[(strip * 2 + nhalf) * 16 + khalf + j] = sums[j];
    }
  }
  __syncthreads();

  // Combine the two N-half partial sums -> per-row normalizer
  float inv[8];
#pragma unroll
  for (int j = 0; j < 8; ++j) {
    float s = sum_lds[strip * 32 + khalf + j] + sum_lds[strip * 32 + 16 + khalf + j];
    inv[j] = 1.0f / fmaxf(s, 1e-6f);
  }

  // ---- GEMM2: msg = (qg @ W) * inv[row]  (this wave: 32 of 64 d-tiles) ----
  const __bf16* a2_base = qg_lds + (strip * 16 + lrow) * QPITCH + khalf;
  const __bf16* wt_base = WTb + (size_t)bb * D_ * G_ + (size_t)lrow * G_ + khalf;
  float* out_base = out + ((size_t)bb * ROWS + (size_t)mt * M_TILE + strip * 16 + khalf) * D_ + lrow;

  for (int i = 0; i < 32; ++i) {
    const int dt = nhalf * 32 + i;
    const __bf16* b_base = wt_base + (size_t)(dt * 16) * G_;
    v8f acc = {0.f, 0.f, 0.f, 0.f, 0.f, 0.f, 0.f, 0.f};
#pragma unroll 4
    for (int ks = 0; ks < G_ / 32; ++ks) {
      const __bf16* ap = a2_base + ks * 32;
      const __bf16* bp = b_base + ks * 32;
      v8bf alo = *(const v8bf*)(ap);
      v8bf ahi = *(const v8bf*)(ap + 16);
      v8bf blo = *(const v8bf*)(bp);
      v8bf bhi = *(const v8bf*)(bp + 16);
      v16bf Am = __builtin_shufflevector(alo, ahi, 0,1,2,3,4,5,6,7,8,9,10,11,12,13,14,15);
      v16bf Bm = __builtin_shufflevector(blo, bhi, 0,1,2,3,4,5,6,7,8,9,10,11,12,13,14,15);
      acc = __builtin_amdgcn_wmma_f32_16x16x32_bf16(false, Am, false, Bm,
                                                    (short)0, acc, false, false);
    }
    float* od = out_base + dt * 16;
#pragma unroll
    for (int j = 0; j < 8; ++j)
      od[(size_t)j * D_] = acc[j] * inv[j];
  }
}

// ---------------------------------------------------------------------------
extern "C" void kernel_launch(void* const* d_in, const int* in_sizes, int n_in,
                              void* d_out, int out_size, void* d_ws, size_t ws_size,
                              hipStream_t stream) {
  const float* qz = (const float*)d_in[0];        // [8,16,512,1024] f32
  const float* W  = (const float*)d_in[1];        // [8,512,1024] f32
  float* out = (float*)d_out;                     // [8,16,512,1024] f32

  __bf16* Wb  = (__bf16*)d_ws;                        // 8 MB
  __bf16* WTb = Wb + (size_t)B_ * G_ * D_;            // 8 MB

  int n = B_ * G_ * D_;
  prep_weights<<<(n + 255) / 256, 256, 0, stream>>>(W, Wb, WTb);

  dim3 grid(ROWS / M_TILE, B_);
  size_t lds_bytes = (size_t)(M_TILE * APITCH + M_TILE * QPITCH) * sizeof(__bf16)
                   + 4 * 2 * 16 * sizeof(float);      // ~195 KB
  topic_fused<<<grid, THREADS, lds_bytes, stream>>>(qz, Wb, WTb, out);
}
